// TransformerEncoder_59253368815811
// MI455X (gfx1250) — compile-verified, matrix operand order checked by reading, not run on hardware
//
#include <hip/hip_runtime.h>
#include <math.h>

// ---------------- problem constants (from reference) ----------------
#define NL   6
#define Hh   16
#define Dd   1024
#define DKk  64
#define DFFf 4096
#define Tt   512
#define Bb   8
#define Mm   (Bb*Tt)          // 4096 rows of the activation matrix
#define NEGP (-3.0e38f)       // mask penalty (matches finfo(f32).min semantics)

// ---------------- WMMA types ----------------
typedef __attribute__((ext_vector_type(16))) __bf16 bf16x16;
typedef __attribute__((ext_vector_type(8)))  float  f32x8;

union BFrag { bf16x16 v; unsigned short u[16]; };

__device__ __forceinline__ f32x8 wmma_bf(const BFrag& a, const BFrag& b, f32x8 c) {
    return __builtin_amdgcn_wmma_f32_16x16x32_bf16(false, a.v, false, b.v,
                                                   (short)0, c, false, false);
}

// round-to-nearest-even f32 -> bf16 bits
__device__ __forceinline__ unsigned short f2bf(float f) {
    unsigned u = __float_as_uint(f);
    u += 0x7FFFu + ((u >> 16) & 1u);
    return (unsigned short)(u >> 16);
}

// K-index mapping for 16-bit A/B fragments (ISA 7.12.2, 16-bit 16x32):
// per lane the 16 elements cover K in two contiguous 8-runs:
//   half=0: {0..7, 16..23}   half=1: {8..15, 24..31}
__device__ __forceinline__ int abk(int j, int half) {
    int v = j >> 1;
    return ((v & 4) << 2) + (half << 3) + ((v & 3) << 1) + (j & 1);
}

// ---------------- prep kernels ----------------
// x [T,B,D] f32 -> xf32/xb [B*T, D]
__global__ void prep_x_k(const float* __restrict__ x,
                         float* __restrict__ xf, unsigned short* __restrict__ xb) {
    int i = blockIdx.x * 256 + threadIdx.x;
    if (i < Tt * Bb * Dd) {
        int d = i & (Dd - 1);
        int rest = i >> 10;
        int b = rest & (Bb - 1);
        int t = rest >> 3;
        float v = x[i];
        size_t o = (size_t)(b * Tt + t) * Dd + d;
        xf[o] = v;
        xb[o] = f2bf(v);
    }
}

// bool padding mask [B,T] -> additive penalty (0 = valid token, NEGP = padded)
__global__ void prep_mask_k(const unsigned char* __restrict__ m,
                            float* __restrict__ vm) {
    int i = blockIdx.x * 256 + threadIdx.x;
    if (i < Bb * Tt) vm[i] = m[i] ? NEGP : 0.0f;
}

// xf32 [B*T,D] -> out [T,B,D]
__global__ void finalize_k(const float* __restrict__ xf, float* __restrict__ out) {
    int i = blockIdx.x * 256 + threadIdx.x;
    if (i < Tt * Bb * Dd) {
        int d = i & (Dd - 1);
        int rest = i >> 10;
        int b = rest & (Bb - 1);
        int t = rest >> 3;
        out[i] = xf[(size_t)(b * Tt + t) * Dd + d];
    }
}

// transpose + convert: in f32 [R][C] (batched), out bf16 [C][R]
// 32x32 LDS tiles, coalesced read and write. R,C multiples of 32.
__global__ __launch_bounds__(256)
void tcvt_k(const float* __restrict__ in, unsigned short* __restrict__ out,
            int R, int C) {
    __shared__ float t[32][33];
    const float* src = in + (size_t)blockIdx.z * R * C;
    unsigned short* dst = out + (size_t)blockIdx.z * R * C;
    int tx = threadIdx.x & 31, ty = threadIdx.x >> 5;   // 32 x 8
#pragma unroll
    for (int i = 0; i < 4; i++) {
        int r = blockIdx.y * 32 + ty + i * 8;
        int c = blockIdx.x * 32 + tx;
        t[ty + i * 8][tx] = src[(size_t)r * C + c];
    }
    __syncthreads();
#pragma unroll
    for (int i = 0; i < 4; i++) {
        int c = blockIdx.x * 32 + ty + i * 8;
        int r = blockIdx.y * 32 + tx;
        dst[(size_t)c * R + r] = f2bf(t[tx][ty + i * 8]);
    }
}

// ---------------- tiled WMMA GEMM with fused epilogues ----------------
// C[M,N] = A(bf16 MxK row-major) * Bt(bf16 NxK row-major, i.e. B transposed)
// Both LDS tiles are [128][TK+8] with identical padded layout so ALL fragment
// reads are two contiguous ds_load_b128 per fragment. TK=64: 16 wmma + 24
// ds_load_b128 per barrier pair.
// MODE 0: store bf16 packed as [b,h,t,dk]            (Q/K projection)
// MODE 4: store bf16 packed as [b,h,dk,t]            (V projection, transposed)
// MODE 1: r=(acc+xf32)/2; update xf32 + xb           (Wo + residual avg)
// MODE 2: gelu_exact(acc+bias) -> bf16 Cout          (FFN1)
// MODE 3: r=((acc+bias)+xf32)/2; update xf32 + xb    (FFN2 + residual avg)
#define TM 128
#define TN 128
#define TK 64
#define LAS (TK + 8)     // 144B row stride: 16B aligned; 36-dword step -> conflict-free

template<int MODE>
__global__ __launch_bounds__(256)
void gemm_bf16_k(const unsigned short* __restrict__ A, int K,
                 const unsigned short* __restrict__ Bt,
                 const float* __restrict__ bias,
                 float* __restrict__ xf32, unsigned short* __restrict__ xbf,
                 unsigned short* __restrict__ Cout, int N) {
    __shared__ unsigned short As[TM * LAS];
    __shared__ unsigned short Bs[TN * LAS];

    const int tid = threadIdx.x, lane = tid & 31, w = tid >> 5;
    const int wm = w >> 2, wn = w & 3;          // 2x4 wave grid
    const int m0 = blockIdx.y * TM, n0 = blockIdx.x * TN;
    const int half = lane >> 4, l16 = lane & 15;

    f32x8 acc[4][2];
#pragma unroll
    for (int i = 0; i < 4; i++)
#pragma unroll
        for (int j = 0; j < 2; j++)
#pragma unroll
            for (int r = 0; r < 8; r++) acc[i][j][r] = 0.0f;

    for (int kt = 0; kt < K; kt += TK) {
        // stage A and Bt tiles (each 128 rows x 64 bf16) as 16B vectors
#pragma unroll
        for (int i = 0; i < 4; i++) {
            int ch = tid + i * 256;             // 1024 chunks of 8 bf16
            int r = ch >> 3, c8 = (ch & 7) * 8;
            *(uint4*)&As[r * LAS + c8] =
                *(const uint4*)&A[(size_t)(m0 + r) * K + kt + c8];
            *(uint4*)&Bs[r * LAS + c8] =
                *(const uint4*)&Bt[(size_t)(n0 + r) * K + kt + c8];
        }
        if (kt + TK < K) {                       // prefetch next tiles
            __builtin_prefetch(&A[(size_t)(m0 + (tid >> 1)) * K + kt + TK], 0, 1);
            __builtin_prefetch(&Bt[(size_t)(n0 + (tid >> 1)) * K + kt + TK], 0, 1);
        }
        __syncthreads();

#pragma unroll
        for (int sub = 0; sub < 2; sub++) {      // two 32-wide K sub-steps
            const int kof = sub * 32;
            BFrag af[4], bf_[2];
#pragma unroll
            for (int i = 0; i < 4; i++) {
                int row = wm * 64 + i * 16 + l16;
#pragma unroll
                for (int j = 0; j < 16; j++)
                    af[i].u[j] = As[row * LAS + kof + abk(j, half)];
            }
#pragma unroll
            for (int i = 0; i < 2; i++) {
                int col = wn * 32 + i * 16 + l16;
#pragma unroll
                for (int j = 0; j < 16; j++)
                    bf_[i].u[j] = Bs[col * LAS + kof + abk(j, half)];
            }
#pragma unroll
            for (int i = 0; i < 4; i++)
#pragma unroll
                for (int j = 0; j < 2; j++)
                    acc[i][j] = wmma_bf(af[i], bf_[j], acc[i][j]);
        }
        __syncthreads();
    }

    // epilogue (C frag: n = lane%16, row m = r + 8*(lane/16))
#pragma unroll
    for (int i = 0; i < 4; i++)
#pragma unroll
        for (int j = 0; j < 2; j++)
#pragma unroll
            for (int r = 0; r < 8; r++) {
                int gm = m0 + wm * 64 + i * 16 + r + (half << 3);
                int gn = n0 + wn * 32 + j * 16 + l16;
                float v = acc[i][j][r];
                if (MODE == 0) {
                    int b = gm >> 9, t = gm & (Tt - 1);
                    int h = gn >> 6, e = gn & (DKk - 1);
                    Cout[(size_t)(((b * Hh + h) * Tt) + t) * DKk + e] = f2bf(v);
                } else if (MODE == 4) {
                    int b = gm >> 9, t = gm & (Tt - 1);
                    int h = gn >> 6, e = gn & (DKk - 1);
                    Cout[(size_t)(((b * Hh + h) * DKk) + e) * Tt + t] = f2bf(v);
                } else if (MODE == 1) {
                    size_t idx = (size_t)gm * Dd + gn;
                    float rr = (v + xf32[idx]) * 0.5f;
                    xf32[idx] = rr;
                    xbf[idx] = f2bf(rr);
                } else if (MODE == 2) {
                    float t = v + bias[gn];
                    float g = 0.5f * t * (1.0f + erff(t * 0.70710678118654752f));
                    Cout[(size_t)gm * N + gn] = f2bf(g);
                } else {
                    float t = v + bias[gn];
                    size_t idx = (size_t)gm * Dd + gn;
                    float rr = (t + xf32[idx]) * 0.5f;
                    xf32[idx] = rr;
                    xbf[idx] = f2bf(rr);
                }
            }
}

// ---------------- fused flash attention (no score materialization) ----------------
// grid: B*H*4 blocks; block = 256 threads (8 waves); wave = 16 query rows.
// Q,K: bf16 [b,h,t,dk]; V: bf16 [b,h,dk,t] (pre-transposed by MODE 4);
// O: bf16 [b*T+t, h*64+dk].
// Scores = Q*K^T + min(qpen,kpen) (NO 1/sqrt(dk), faithful to the reference);
// online softmax in f32 (row max via 16-lane butterflies; row SUM via an extra
// wmma(P, ones) on the matrix pipe — zero cross-lane ops, and the normalizer
// sums exactly the bf16-rounded P used in the PV product);
// P relayout C->A via per-wave LDS bounce; PV via wmma.
__global__ __launch_bounds__(256)
void attn_k(const unsigned short* __restrict__ Q,
            const unsigned short* __restrict__ Kb,
            const unsigned short* __restrict__ Vt,
            const float* __restrict__ vmask,
            unsigned short* __restrict__ O) {
    __shared__ unsigned short Ks[128 * 72];     // [key][dk]  : frag K-runs contiguous
    __shared__ unsigned short Vs[64 * 136];     // [dk][key]  : frag K-runs contiguous
    __shared__ unsigned short Ps[8 * 16 * 40];  // per-wave P bounce (16x32, padded)

    const int tid = threadIdx.x, lane = tid & 31, w = tid >> 5;
    const int half = lane >> 4, l16 = lane & 15;
    const int bh = blockIdx.x >> 2, qc = blockIdx.x & 3;
    const int b = bh >> 4, h = bh & (Hh - 1);
    const size_t base = (size_t)bh * Tt * DKk;
    const int q0 = qc * 128 + w * 16;

    // Q A-fragments (dk 0..31 and 32..63), resident for all key blocks
    BFrag qa[2];
    {
        int qrow = q0 + l16;
#pragma unroll
        for (int f = 0; f < 2; f++)
#pragma unroll
            for (int j = 0; j < 16; j++)
                qa[f].u[j] = Q[base + (size_t)qrow * DKk + f * 32 + abk(j, half)];
    }
    float qp[8];                                 // per-row additive penalty (0 / NEGP)
#pragma unroll
    for (int r = 0; r < 8; r++)
        qp[r] = vmask[b * Tt + q0 + r + (half << 3)];

    BFrag ones;                                  // all-1.0 bf16 B fragment
#pragma unroll
    for (int j = 0; j < 16; j++) ones.u[j] = 0x3F80u;

    f32x8 o[4];
#pragma unroll
    for (int i = 0; i < 4; i++)
#pragma unroll
        for (int r = 0; r < 8; r++) o[i][r] = 0.0f;
    float rmax[8], rsum[8];
#pragma unroll
    for (int r = 0; r < 8; r++) { rmax[r] = NEGP; rsum[r] = 0.0f; }

    unsigned short* Pw = &Ps[w * 16 * 40];

    for (int cc = 0; cc < 4; ++cc) {            // 128-key chunks
        __syncthreads();
#pragma unroll
        for (int i = 0; i < 4; i++) {           // stage K [key][dk] chunk
            int ch = tid + i * 256;             // 1024 chunks of 8
            int kr = ch >> 3, c8 = (ch & 7) * 8;
            *(uint4*)&Ks[kr * 72 + c8] =
                *(const uint4*)&Kb[base + (size_t)(cc * 128 + kr) * DKk + c8];
        }
#pragma unroll
        for (int i = 0; i < 4; i++) {           // stage V^T [dk][key] chunk
            int ch = tid + i * 256;             // 1024 chunks of 8
            int dk = ch >> 4, c8 = (ch & 15) * 8;
            *(uint4*)&Vs[dk * 136 + c8] =
                *(const uint4*)&Vt[base + (size_t)dk * Tt + cc * 128 + c8];
        }
        __syncthreads();

        for (int kb = 0; kb < 4; ++kb) {        // 32-key blocks
            const int key0 = kb * 32;
            f32x8 s[2];
#pragma unroll
            for (int f = 0; f < 2; f++)
#pragma unroll
                for (int r = 0; r < 8; r++) s[f][r] = 0.0f;

#pragma unroll
            for (int f = 0; f < 2; f++) {       // S = Q*K^T over dk (2 halves)
                BFrag k0, k1;
                int key = key0 + f * 16 + l16;
#pragma unroll
                for (int j = 0; j < 16; j++) {
                    k0.u[j] = Ks[key * 72 + abk(j, half)];
                    k1.u[j] = Ks[key * 72 + 32 + abk(j, half)];
                }
                s[f] = wmma_bf(qa[0], k0, s[f]);
                s[f] = wmma_bf(qa[1], k1, s[f]);
            }

            // padding penalty: min(qpen,kpen) is 0 iff both tokens valid
            float kp[2];
#pragma unroll
            for (int f = 0; f < 2; f++)
                kp[f] = vmask[b * Tt + cc * 128 + key0 + f * 16 + l16];
#pragma unroll
            for (int f = 0; f < 2; f++)
#pragma unroll
                for (int r = 0; r < 8; r++)
                    s[f][r] += fminf(qp[r], kp[f]);

            // running row max (must be row-uniform: 4-stage 16-lane butterfly)
            float scale[8];
#pragma unroll
            for (int r = 0; r < 8; r++) {
                float mx = fmaxf(s[0][r], s[1][r]);
#pragma unroll
                for (int off = 1; off < 16; off <<= 1)
                    mx = fmaxf(mx, __shfl_xor(mx, off, 32));
                float mn = fmaxf(rmax[r], mx);
                scale[r] = __expf(rmax[r] - mn);
                rmax[r] = mn;
            }
            // exponentiate (no cross-lane sum here; sum comes from wmma below)
#pragma unroll
            for (int f = 0; f < 2; f++)
#pragma unroll
                for (int r = 0; r < 8; r++)
                    s[f][r] = __expf(s[f][r] - rmax[r]);
            // rescale running output accumulator
#pragma unroll
            for (int i = 0; i < 4; i++)
#pragma unroll
                for (int r = 0; r < 8; r++) o[i][r] *= scale[r];

            // P: C-layout -> row-major LDS -> A-fragment (contiguous runs)
#pragma unroll
            for (int f = 0; f < 2; f++)
#pragma unroll
                for (int r = 0; r < 8; r++)
                    Pw[(r + (half << 3)) * 40 + f * 16 + l16] = f2bf(s[f][r]);
            BFrag pa;
#pragma unroll
            for (int j = 0; j < 16; j++) pa.u[j] = Pw[l16 * 40 + abk(j, half)];

            // row sums of P via the matrix pipe: every column of P*ones = rowsum
            f32x8 sum8;
#pragma unroll
            for (int r = 0; r < 8; r++) sum8[r] = 0.0f;
            sum8 = wmma_bf(pa, ones, sum8);
#pragma unroll
            for (int r = 0; r < 8; r++)
                rsum[r] = rsum[r] * scale[r] + sum8[r];

            // O += P * V   (B frag from V^T: n=dk fixed per lane, k=key runs)
#pragma unroll
            for (int i = 0; i < 4; i++) {
                BFrag vb_;
#pragma unroll
                for (int j = 0; j < 16; j++)
                    vb_.u[j] = Vs[(i * 16 + l16) * 136 + key0 + abk(j, half)];
                o[i] = wmma_bf(pa, vb_, o[i]);
            }
        }
    }

    // normalize and store O[b*T+t][h*64+dk]
#pragma unroll
    for (int i = 0; i < 4; i++)
#pragma unroll
        for (int r = 0; r < 8; r++) {
            int qrow = q0 + r + (half << 3);
            float val = o[i][r] / rsum[r];
            O[(size_t)(b * Tt + qrow) * (Hh * DKk) + h * DKk + i * 16 + l16] =
                f2bf(val);
        }
}

// ---------------- host orchestration ----------------
extern "C" void kernel_launch(void* const* d_in, const int* in_sizes, int n_in,
                              void* d_out, int out_size, void* d_ws, size_t ws_size,
                              hipStream_t stream) {
    const float* x_in  = (const float*)d_in[0];
    const unsigned char* mask = (const unsigned char*)d_in[1];   // jax bool = 1 byte
    const float* Wq = (const float*)d_in[2];
    const float* Wk = (const float*)d_in[3];
    const float* Wv = (const float*)d_in[4];
    const float* Wo = (const float*)d_in[5];
    const float* W1 = (const float*)d_in[6];
    const float* b1 = (const float*)d_in[7];
    const float* W2 = (const float*)d_in[8];
    const float* b2 = (const float*)d_in[9];
    float* out = (float*)d_out;

    // workspace carve-up (256B aligned)
    size_t off = 0;
    char* wsB = (char*)d_ws;
    auto carve = [&](size_t bytes) -> void* {
        void* p = wsB + off;
        off += (bytes + 255) & ~(size_t)255;
        return p;
    };
    float*          xf  = (float*)         carve((size_t)Mm * Dd * 4);
    unsigned short* xb  = (unsigned short*)carve((size_t)Mm * Dd * 2);
    float*          vm  = (float*)         carve((size_t)Bb * Tt * 4);
    unsigned short* qb  = (unsigned short*)carve((size_t)Mm * Dd * 2);
    unsigned short* kb_ = (unsigned short*)carve((size_t)Mm * Dd * 2);
    unsigned short* vb_ = (unsigned short*)carve((size_t)Mm * Dd * 2);
    unsigned short* ob  = (unsigned short*)carve((size_t)Mm * Dd * 2);
    unsigned short* fb  = (unsigned short*)carve((size_t)Mm * DFFf * 2);
    unsigned short* wq  = (unsigned short*)carve((size_t)Hh * Dd * DKk * 2);
    unsigned short* wk  = (unsigned short*)carve((size_t)Hh * Dd * DKk * 2);
    unsigned short* wv  = (unsigned short*)carve((size_t)Hh * Dd * DKk * 2);
    unsigned short* wo  = (unsigned short*)carve((size_t)Dd * Dd * 2);
    unsigned short* w1  = (unsigned short*)carve((size_t)Dd * DFFf * 2);
    unsigned short* w2  = (unsigned short*)carve((size_t)DFFf * Dd * 2);

    const int nx = Tt * Bb * Dd;
    prep_x_k<<<(nx + 255) / 256, 256, 0, stream>>>(x_in, xf, xb);
    prep_mask_k<<<(Bb * Tt + 255) / 256, 256, 0, stream>>>(mask, vm);

    const int wqkv_n = Hh * Dd * DKk;      // 1,048,576
    const int wff_n  = Dd * DFFf;          // 4,194,304
    const dim3 blk(256);
    const dim3 gD(Dd / TN, Mm / TM);       // N=1024 GEMMs: 8 x 32
    const dim3 gF(DFFf / TN, Mm / TM);     // N=4096 GEMM : 32 x 32
    // weight transpose grids: (C/32, R/32, batch) for in f32 [R][C] -> out bf16 [C][R]
    const dim3 gQKV(DKk / 32, Dd / 32, Hh);     // per-head 1024x64 -> [h][e][d]
    const dim3 gWo(Dd / 32, Dd / 32, 1);
    const dim3 gW1(DFFf / 32, Dd / 32, 1);
    const dim3 gW2(Dd / 32, DFFf / 32, 1);

    for (int i = 0; i < NL; ++i) {
        // per-layer weight transpose-convert f32 -> bf16 [N][K]
        tcvt_k<<<gQKV, blk, 0, stream>>>(Wq + (size_t)i * wqkv_n, wq, Dd, DKk);
        tcvt_k<<<gQKV, blk, 0, stream>>>(Wk + (size_t)i * wqkv_n, wk, Dd, DKk);
        tcvt_k<<<gQKV, blk, 0, stream>>>(Wv + (size_t)i * wqkv_n, wv, Dd, DKk);
        tcvt_k<<<gWo,  blk, 0, stream>>>(Wo + (size_t)i * (Dd * Dd), wo, Dd, Dd);
        tcvt_k<<<gW1,  blk, 0, stream>>>(W1 + (size_t)i * wff_n, w1, Dd, DFFf);
        tcvt_k<<<gW2,  blk, 0, stream>>>(W2 + (size_t)i * wff_n, w2, DFFf, Dd);

        // Q/K projections -> [b,h,t,dk]; V projection -> [b,h,dk,t]
        gemm_bf16_k<0><<<gD, blk, 0, stream>>>(xb, Dd, wq, nullptr, nullptr, nullptr, qb, Dd);
        gemm_bf16_k<0><<<gD, blk, 0, stream>>>(xb, Dd, wk, nullptr, nullptr, nullptr, kb_, Dd);
        gemm_bf16_k<4><<<gD, blk, 0, stream>>>(xb, Dd, wv, nullptr, nullptr, nullptr, vb_, Dd);

        // fused attention: B*H*(T/128) = 512 workgroups
        attn_k<<<Bb * Hh * 4, blk, 0, stream>>>(qb, kb_, vb_, vm, ob);

        // x = (o*Wo + x)/2
        gemm_bf16_k<1><<<gD, blk, 0, stream>>>(ob, Dd, wo, nullptr, xf, xb, nullptr, Dd);
        // f = gelu(x*W1 + b1)
        gemm_bf16_k<2><<<gF, blk, 0, stream>>>(xb, Dd, w1, b1 + (size_t)i * DFFf,
                                               nullptr, nullptr, fb, DFFf);
        // x = ((f*W2 + b2) + x)/2
        gemm_bf16_k<3><<<gD, blk, 0, stream>>>(fb, DFFf, w2, b2 + (size_t)i * Dd,
                                               xf, xb, nullptr, Dd);
    }

    finalize_k<<<(nx + 255) / 256, 256, 0, stream>>>(xf, out);
}
